// VectorQuantizer_35974646071746
// MI455X (gfx1250) — matrix operand order repaired and checked by reading, not compiled, not run on hardware
//
#include <hip/hip_runtime.h>
#include <float.h>

typedef __bf16 v16bf __attribute__((ext_vector_type(16)));
typedef __bf16 v4bf  __attribute__((ext_vector_type(4)));
typedef float  v8f   __attribute__((ext_vector_type(8)));

#define B_ 4
#define H_ 8
#define R_ 8
#define C_ 512
#define K_ 128
#define S_ 1024
// rows per head = B*R*C = 16384 ; total rows = 131072 ; row tiles = 8192

// d_out layout (floats), reference return order:
// quantized [B,H,R,C,K], z [B,H,R,C], l_commit, l_codebook, errs2 [B,H,R,C]
#define QOFF    0
#define ZOFF    16777216u          // B*H*R*C*K
#define NROWS   131072u            // B*H*R*C
#define LCOMMIT (ZOFF + NROWS)     // 16908288
#define LCODE   (LCOMMIT + 1u)
#define EOFF    (LCODE + 1u)

// -------------------------------------------------------------------------
// Kernel 1: codebook prep. One wave per codeword (H*S = 8192 codewords).
// c_f32 = c_sum / max(count, 0.01); also bf16 copy and ||c||^2.
// -------------------------------------------------------------------------
__global__ void __launch_bounds__(256)
vq_codebook_kernel(const float* __restrict__ c_sum,
                   const float* __restrict__ c_count,
                   float* __restrict__ c_f32,
                   __bf16* __restrict__ c_bf16,
                   float* __restrict__ cnorm) {
    const int cw   = blockIdx.x * 8 + (threadIdx.x >> 5);  // 0..8191
    const int lane = threadIdx.x & 31;

    const float inv = 1.0f / fmaxf(c_count[cw], 0.01f);
    float4 v = ((const float4*)(c_sum + (size_t)cw * K_))[lane];
    v.x *= inv; v.y *= inv; v.z *= inv; v.w *= inv;

    ((float4*)(c_f32 + (size_t)cw * K_))[lane] = v;

    v4bf b;
    b[0] = (__bf16)v.x; b[1] = (__bf16)v.y; b[2] = (__bf16)v.z; b[3] = (__bf16)v.w;
    ((v4bf*)(c_bf16 + (size_t)cw * K_))[lane] = b;

    float sq = v.x * v.x + v.y * v.y + v.z * v.z + v.w * v.w;
#pragma unroll
    for (int off = 16; off; off >>= 1) sq += __shfl_xor(sq, off, 32);
    if (lane == 0) cnorm[cw] = sq;
}

// -------------------------------------------------------------------------
// Kernel 2: distance matmul + argmin.
// Block = 8 waves, all in the same head; each wave owns one 16-row tile.
// The 16-codeword (16x128 bf16 = 4KB) B tile is staged in LDS, double
// buffered: global loads for tile st+1 issue before the WMMAs of tile st,
// ds_store lands after compute, one barrier per tile.
// dist(n,s) = ||v_n||^2 - 2 v_n.c_s + ||c_s||^2 ; -2*dot term via
// v_wmma_f32_16x16x32_bf16 over K=128 (4 chained WMMAs per 16x16 tile).
// -------------------------------------------------------------------------
__global__ void __launch_bounds__(256)
vq_assign_kernel(const float* __restrict__ vecs,
                 const __bf16* __restrict__ c_bf16,
                 const float* __restrict__ cnorm,
                 float* __restrict__ out,
                 int* __restrict__ zint,
                 float* __restrict__ partials) {
    __shared__ __align__(32) __bf16 lbuf[2][16 * K_];  // 2 x 4KB codebook tiles
    __shared__ float lcn[S_];                          // 4KB: ||c||^2 for head

    const int tid  = threadIdx.x;
    const int wid  = tid >> 5;
    const int lane = tid & 31;

    const int h    = blockIdx.x >> 7;                    // 128 blocks per head
    const int tih  = ((blockIdx.x & 127) << 3) + wid;    // tile within head
    const int wave = (h << 10) + tih;                    // global tile id
    const int n0   = tih << 4;                           // first row in head
    const int b    = n0 >> 12;                           // 4096 rows per (b,h)
    const int m0   = n0 & 4095;                          // r*C + c part

    // ---- stage ||c||^2 of this head into LDS (1024 floats, 4 per thread)
    {
        const float4 t = ((const float4*)(cnorm + h * S_))[tid];
        ((float4*)lcn)[tid] = t;
    }

    // ---- load A tile (16 rows x 128) once; build bf16 fragments in contiguous-K
    // layout: lane<16 holds K {0..15}+32*ks of row lane; lane>=16 holds
    // K {16..31}+32*ks of row lane-16. Accumulate ||v||^2 on the fly.
    const float* vp = vecs + ((size_t)(b * H_ + h) * 4096 + m0) * K_
                           + (size_t)(lane & 15) * K_;
    const int kb = (lane >> 4) << 4;   // 0 or 16

    v16bf a[4];
    float sumsq = 0.0f;
#pragma unroll
    for (int ks = 0; ks < 4; ++ks) {
        const float* p = vp + ks * 32 + kb;
        const float4 x0 = *(const float4*)(p);
        const float4 x1 = *(const float4*)(p + 4);
        const float4 x2 = *(const float4*)(p + 8);
        const float4 x3 = *(const float4*)(p + 12);
        a[ks][0]  = (__bf16)x0.x; a[ks][1]  = (__bf16)x0.y;
        a[ks][2]  = (__bf16)x0.z; a[ks][3]  = (__bf16)x0.w;
        a[ks][4]  = (__bf16)x1.x; a[ks][5]  = (__bf16)x1.y;
        a[ks][6]  = (__bf16)x1.z; a[ks][7]  = (__bf16)x1.w;
        a[ks][8]  = (__bf16)x2.x; a[ks][9]  = (__bf16)x2.y;
        a[ks][10] = (__bf16)x2.z; a[ks][11] = (__bf16)x2.w;
        a[ks][12] = (__bf16)x3.x; a[ks][13] = (__bf16)x3.y;
        a[ks][14] = (__bf16)x3.z; a[ks][15] = (__bf16)x3.w;
        sumsq += x0.x*x0.x + x0.y*x0.y + x0.z*x0.z + x0.w*x0.w
               + x1.x*x1.x + x1.y*x1.y + x1.z*x1.z + x1.w*x1.w
               + x2.x*x2.x + x2.y*x2.y + x2.z*x2.z + x2.w*x2.w
               + x3.x*x3.x + x3.y*x3.y + x3.z*x3.z + x3.w*x3.w;
    }

    const __bf16* cb = c_bf16 + (size_t)h * S_ * K_;

    float minv[8];
    int   mini[8];
#pragma unroll
    for (int j = 0; j < 8; ++j) { minv[j] = FLT_MAX; mini[j] = 0; }

    const int sl   = lane & 15;          // codeword within tile (B column)
    const int koff = (lane >> 4) << 4;   // 0 or 16: K half for B fragment

    // ---- preload codebook tile 0 into LDS (256 threads x 16B = 4KB)
    {
        const float4 t0 = ((const float4*)cb)[tid];
        ((float4*)lbuf[0])[tid] = t0;
    }
    __syncthreads();

    for (int st = 0; st < 64; ++st) {
        const int s0 = st << 4;

        // issue global loads for the next tile early (latency hidden by WMMA)
        float4 nxt;
        if (st < 63) {
            nxt = ((const float4*)(cb + (size_t)(s0 + 16) * K_))[tid];
            if (st < 62)
                __builtin_prefetch(cb + (size_t)(s0 + 32) * K_ + (size_t)tid * 8, 0, 1);
        }

        const __bf16* bbase = lbuf[st & 1] + sl * K_ + koff;
        v8f acc = {};
#pragma unroll
        for (int ks = 0; ks < 4; ++ks) {
            const v16bf bfrag = *(const v16bf*)(bbase + ks * 32);
            acc = __builtin_amdgcn_wmma_f32_16x16x32_bf16(
                false, a[ks], false, bfrag, (short)0, acc, false, false);
        }

        const float cnv  = lcn[s0 + sl];
        const int   sidx = s0 + sl;
#pragma unroll
        for (int j = 0; j < 8; ++j) {
            const float pv = cnv - 2.0f * acc[j];
            if (pv < minv[j]) { minv[j] = pv; mini[j] = sidx; }
        }

        // park the next tile into the other buffer (its last readers
        // finished before the barrier that ended iteration st-1)
        if (st < 63) ((float4*)lbuf[(st + 1) & 1])[tid] = nxt;
        __syncthreads();
    }

    // ---- epilogue: cross-lane argmin within each 16-lane group.
    // acc VGPR j: lanes 0-15 -> row j, lanes 16-31 -> row j+8.
    const float tot = sumsq + __shfl_xor(sumsq, 16, 32);  // ||v||^2 of row lane&15
    const int   grp = lane >> 4;
    const size_t gidbase = (size_t)(b * H_ + h) * 4096 + m0;

    float commit_local = 0.0f;
#pragma unroll
    for (int j = 0; j < 8; ++j) {
        float val = minv[j];
        int   idx = mini[j];
#pragma unroll
        for (int off = 8; off; off >>= 1) {
            const float ov = __shfl_xor(val, off, 32);
            const int   oi = __shfl_xor(idx, off, 32);
            if (ov < val || (ov == val && oi < idx)) { val = ov; idx = oi; }
        }
        const int m = j + (grp << 3);
        const float vn = __shfl(tot, m, 32);  // lane m (0..15) holds row m norm
        if (sl == 0) {
            const float err = fmaxf(vn + val, 0.0f);
            const size_t gid = gidbase + m;
            out[ZOFF + gid] = (float)idx;
            out[EOFF + gid] = err;
            zint[gid] = idx;
            commit_local += err;
        }
    }
    const float pr = commit_local + __shfl_xor(commit_local, 16, 32);
    if (lane == 0) partials[wave] = pr;
}

// -------------------------------------------------------------------------
// Kernel 3: gather quantized[row,:] = c_f32[h, z[row], :]  (pure bandwidth)
// -------------------------------------------------------------------------
__global__ void __launch_bounds__(256)
vq_gather_kernel(const float* __restrict__ c_f32,
                 const int* __restrict__ zint,
                 float* __restrict__ qout) {
    const int gid  = blockIdx.x * 8 + (threadIdx.x >> 5);  // 0..131071
    const int lane = threadIdx.x & 31;
    const int h = (gid >> 12) & 7;
    const int z = zint[gid];
    const float4* src = (const float4*)(c_f32 + ((size_t)(h * S_ + z)) * K_);
    float4* dst = (float4*)(qout + (size_t)gid * K_);
    dst[lane] = src[lane];
}

// -------------------------------------------------------------------------
// Kernel 4: deterministic reduction of per-tile commit sums; l_codebook = 0
// (forward value of st(x) is exactly zero).
// -------------------------------------------------------------------------
__global__ void __launch_bounds__(256)
vq_reduce_kernel(const float* __restrict__ partials, float* __restrict__ out) {
    __shared__ float sm[256];
    const int tid = threadIdx.x;
    float s = 0.0f;
    for (int i = tid; i < 8192; i += 256) s += partials[i];
    sm[tid] = s;
    __syncthreads();
    for (int w = 128; w; w >>= 1) {
        if (tid < w) sm[tid] += sm[tid + w];
        __syncthreads();
    }
    if (tid == 0) {
        out[LCOMMIT] = sm[0] * (1.0f / 16384.0f);  // mean over B*R*C
        out[LCODE]   = 0.0f;
    }
}

// -------------------------------------------------------------------------
extern "C" void kernel_launch(void* const* d_in, const int* in_sizes, int n_in,
                              void* d_out, int out_size, void* d_ws, size_t ws_size,
                              hipStream_t stream) {
    const float* vecs    = (const float*)d_in[0];  // [B,H,R,C,K] fp32
    const float* c_sum   = (const float*)d_in[1];  // [H,S,K] fp32
    const float* c_count = (const float*)d_in[2];  // [H,S] fp32
    float* out = (float*)d_out;

    // workspace carve-up
    char* ws = (char*)d_ws;
    float*  c_f32    = (float*)(ws);                               // 4 MB
    __bf16* cbf      = (__bf16*)(ws + (4u << 20));                 // 2 MB
    float*  cnorm    = (float*)(ws + (6u << 20));                  // 32 KB
    int*    zint     = (int*)(ws + (6u << 20) + (32u << 10));      // 512 KB
    float*  partials = (float*)(ws + (6u << 20) + (544u << 10));   // 32 KB

    vq_codebook_kernel<<<1024, 256, 0, stream>>>(c_sum, c_count, c_f32, cbf, cnorm);
    vq_assign_kernel  <<<1024, 256, 0, stream>>>(vecs, cbf, cnorm, out, zint, partials);
    vq_gather_kernel  <<<16384, 256, 0, stream>>>(c_f32, zint, out + QOFF);
    vq_reduce_kernel  <<<1, 256, 0, stream>>>(partials, out);
}
